// Mamba2PeptideClassifier_70574902608642
// MI455X (gfx1250) — matrix operand order repaired and checked by reading, not compiled, not run on hardware
//
#include <hip/hip_runtime.h>
#include <hip/hip_bf16.h>
#include <math.h>

#define BATCH   8
#define SEQLEN  512
#define NT      (BATCH*SEQLEN)          // 4096 tokens
#define DMODEL  512
#define DINNER  1024
#define NHEADS  16
#define HEADDIM 64
#define DSTATE  128
#define CONVDIM 1280                    // DINNER + 2*DSTATE
#define DINPROJ 2320                    // 2*DINNER + 2*DSTATE + NHEADS
#define CHUNK   64
#define NCHUNK  8
#define NLAYER  4
#define RMS_EPS 1e-5f

#if defined(__has_builtin)
#if __has_builtin(__builtin_amdgcn_global_load_async_to_lds_b128)
#define HAS_ASYNC_LDS 1
#endif
#if __has_builtin(__builtin_amdgcn_s_wait_asynccnt)
#define HAS_WAIT_ASYNC_BUILTIN 1
#endif
#endif
#ifndef HAS_ASYNC_LDS
#define HAS_ASYNC_LDS 0
#endif
#ifndef HAS_WAIT_ASYNC_BUILTIN
#define HAS_WAIT_ASYNC_BUILTIN 0
#endif

#define AS1 __attribute__((address_space(1)))
#define AS3 __attribute__((address_space(3)))

typedef __bf16 bf16;
typedef __attribute__((ext_vector_type(16))) __bf16 v16bf;
typedef __attribute__((ext_vector_type(8)))  __bf16 v8bf;
typedef __attribute__((ext_vector_type(8)))  float  v8f;
typedef __attribute__((ext_vector_type(2)))  float  v2f;
typedef __attribute__((ext_vector_type(4)))  int    v4i;

__device__ __forceinline__ bf16 f2bf(float f) {
  unsigned u = __builtin_bit_cast(unsigned, f);
  unsigned r = u + 0x7FFFu + ((u >> 16) & 1u);   // round-to-nearest-even
  unsigned short h = (unsigned short)(r >> 16);
  return __builtin_bit_cast(bf16, h);
}

// A fragment 16x32 bf16: lane holds row m, elems K[off..off+7] and K[off+16..off+23], off=8*half.
// Pass p = &A[m*lda + k0 + 8*half]  (works for global or LDS rows of stride >= 32).
__device__ __forceinline__ v16bf load_a16(const bf16* p) {
  v8bf lo = *(const v8bf*)(p);
  v8bf hi = *(const v8bf*)(p + 16);
  return __builtin_shufflevector(lo, hi, 0,1,2,3,4,5,6,7,8,9,10,11,12,13,14,15);
}
// B fragment 32x16 bf16: lane holds column n, elems K[off..off+7] and K[off+16..off+23], off=8*half.
// Pass p = &W[n*ldb + k0 + 8*half]  (W stored N x K row-major).
__device__ __forceinline__ v16bf load_b16(const bf16* p) {
  v8bf lo = *(const v8bf*)(p);
  v8bf hi = *(const v8bf*)(p + 16);
  return __builtin_shufflevector(lo, hi, 0,1,2,3,4,5,6,7,8,9,10,11,12,13,14,15);
}

__device__ __forceinline__ void wait_async_zero() {
#if HAS_ASYNC_LDS
#if HAS_WAIT_ASYNC_BUILTIN
  __builtin_amdgcn_s_wait_asynccnt(0);
#else
  asm volatile("s_wait_asynccnt 0x0" ::: "memory");
#endif
#endif
}

#if HAS_ASYNC_LDS
// async 16B global -> LDS copy (ASYNCcnt tracked)
__device__ __forceinline__ void async_copy16(const bf16* gsrc, bf16* ldst) {
  __builtin_amdgcn_global_load_async_to_lds_b128(
      (AS1 v4i*)(AS1 void*)(void*)const_cast<bf16*>(gsrc),
      (AS3 v4i*)(AS3 void*)(void*)ldst,
      0, 0);
}
#endif

// ---------------------------------------------------------------- utilities
__global__ void k_f32_to_bf16(const float* __restrict__ s, bf16* __restrict__ d, int n) {
  int i = blockIdx.x * 256 + threadIdx.x;
  if (i < n) d[i] = f2bf(s[i]);
}

__global__ void k_embed(const int* __restrict__ tok, const float* __restrict__ emb,
                        float* __restrict__ x) {
  int t = blockIdx.x, tid = threadIdx.x;
  int tk = tok[t];
  const float* er = emb + (size_t)tk * DMODEL;
  float* xr = x + (size_t)t * DMODEL;
  xr[tid]       = er[tid];
  xr[tid + 256] = er[tid + 256];
}

// ------------------------------------------------------- rmsnorm -> bf16 (pre-mixer)
__global__ void k_rmsnorm_in(const float* __restrict__ x, const float* __restrict__ w,
                             bf16* __restrict__ o) {
  __shared__ float sb[256];
  int t = blockIdx.x, tid = threadIdx.x;
  const float* xr = x + (size_t)t * DMODEL;
  float v0 = xr[tid], v1 = xr[tid + 256];
  sb[tid] = v0 * v0 + v1 * v1;
  __syncthreads();
  for (int s = 128; s > 0; s >>= 1) { if (tid < s) sb[tid] += sb[tid + s]; __syncthreads(); }
  float ri = rsqrtf(sb[0] / (float)DMODEL + RMS_EPS);
  bf16* orow = o + (size_t)t * DMODEL;
  orow[tid]       = f2bf(v0 * ri * w[tid]);
  orow[tid + 256] = f2bf(v1 * ri * w[tid + 256]);
}

// ------------------------------------------------------- bf16 WMMA GEMM: C = A(MxK) * W(NxK)^T
// Block = 128 threads (4 waves) -> 64(M) x 64(N) tile; wave w does the 64x16 sub-tile.
// A tile (64x32 per k-step) is staged in LDS with async global->LDS copies, double buffered.
__global__ __launch_bounds__(128)
void k_gemm_bf16(const bf16* __restrict__ A, const bf16* __restrict__ W,
                 float* __restrict__ C, int M, int N, int K, int accumulate) {
  __shared__ bf16 smA[2][64 * 32];      // 2 x 4KB
  const int tid  = threadIdx.x;
  const int lane = tid & 31;
  const int wv   = tid >> 5;
  const int r = lane & 15, hf = lane >> 4;
  const int m0 = blockIdx.x * 64;
  int n0 = blockIdx.y * 64 + wv * 16;
  if (n0 > N - 16) n0 = N - 16;         // ragged-N: clamp (duplicate tile, identical writes)

  // staging assignment: 256 16B-chunks per k-step; thread copies chunks tid and tid+128.
  const int c1   = tid + 128;
  const int row0 = tid >> 2, off0 = (tid & 3) * 8;
  const int row1 = c1  >> 2, off1 = (c1  & 3) * 8;
  const bf16* gA0 = A + (size_t)(m0 + row0) * K + off0;
  const bf16* gA1 = A + (size_t)(m0 + row1) * K + off1;
  bf16* sA0[2] = { &smA[0][row0 * 32 + off0], &smA[1][row0 * 32 + off0] };
  bf16* sA1[2] = { &smA[0][row1 * 32 + off1], &smA[1][row1 * 32 + off1] };

  v8f acc[4] = {};
  const bf16* wp = W + (size_t)(n0 + r) * K + hf * 8;

  auto stage = [&](int b, int kk) {
#if HAS_ASYNC_LDS
    async_copy16(gA0 + kk, sA0[b]);
    async_copy16(gA1 + kk, sA1[b]);
#else
    v8bf t0 = *(const v8bf*)(gA0 + kk);
    v8bf t1 = *(const v8bf*)(gA1 + kk);
    *(v8bf*)sA0[b] = t0;
    *(v8bf*)sA1[b] = t1;
#endif
  };

  stage(0, 0);                           // prologue
  int buf = 0;
  for (int k = 0; k < K; k += 32) {
    wait_async_zero();                   // my async writes to smA[buf] done
    __syncthreads();                     // everyone's writes visible; prev reads finished
    if (k + 32 < K) stage(buf ^ 1, k + 32);
    if (k + 128 < K) __builtin_prefetch((const void*)(wp + k + 128), 0, 0);
    v16bf bfrag = load_b16(wp + k);
    const bf16* sa = &smA[buf][r * 32 + hf * 8];
    #pragma unroll
    for (int i = 0; i < 4; ++i) {
      v16bf afrag = load_a16(sa + i * 16 * 32);
      acc[i] = __builtin_amdgcn_wmma_f32_16x16x32_bf16(false, afrag, false, bfrag,
                                                       (short)0, acc[i], false, false);
    }
    buf ^= 1;
  }

  #pragma unroll
  for (int i = 0; i < 4; ++i) {
    #pragma unroll
    for (int j = 0; j < 8; ++j) {
      int row = m0 + i * 16 + hf * 8 + j;
      int col = n0 + r;
      size_t idx = (size_t)row * N + col;
      float v = acc[i][j];
      if (accumulate) v += C[idx];
      C[idx] = v;
    }
  }
}

// ------------------------------------------------------- dt = softplus(.. + bias); dA = -exp(Alog)*dt
__global__ void k_dt(const float* __restrict__ zx, const float* __restrict__ dtb,
                     const float* __restrict__ alog, float* __restrict__ dt,
                     float* __restrict__ dA) {
  int i = blockIdx.x * 256 + threadIdx.x;   // < NT*NHEADS
  if (i >= NT * NHEADS) return;
  int t = i >> 4, h = i & 15;
  float raw = zx[(size_t)t * DINPROJ + DINNER + CONVDIM + h] + dtb[h];
  float d = (raw > 20.0f) ? raw : log1pf(expf(raw));
  dt[i] = d;
  dA[i] = -expf(alog[h]) * d;
}

// ------------------------------------------------------- depthwise causal conv (4 taps) + SiLU
__global__ void k_conv(const float* __restrict__ zx, const float* __restrict__ cw,
                       const float* __restrict__ cb, float* __restrict__ xBC) {
  int t = blockIdx.x;
  int ch = blockIdx.y * 256 + threadIdx.x;
  int b = t / SEQLEN, l = t % SEQLEN;
  float acc = cb[ch];
  #pragma unroll
  for (int k = 0; k < 4; ++k) {
    int ls = l + k - 3;
    if (ls >= 0)
      acc += zx[(size_t)(b * SEQLEN + ls) * DINPROJ + DINNER + ch] * cw[ch * 4 + k];
  }
  xBC[(size_t)t * CONVDIM + ch] = acc / (1.0f + expf(-acc));
}

__global__ void k_xdt(const float* __restrict__ xBC, const float* __restrict__ dt,
                      float* __restrict__ xdt) {
  int i = blockIdx.x * 256 + threadIdx.x;   // NT*DINNER
  int t = i >> 10, ch = i & 1023, h = ch >> 6;
  xdt[i] = xBC[(size_t)t * CONVDIM + ch] * dt[t * NHEADS + h];
}

// per (b,c,h): inclusive cumsum of dA over l inside chunk -> Acs[(bc*16+h)*64 + l]
__global__ void k_cumsum(const float* __restrict__ dA, float* __restrict__ Acs) {
  int e = blockIdx.x * 256 + threadIdx.x;   // < B*NC*H = 1024
  int bc = e >> 4, h = e & 15;
  float a = 0.0f;
  for (int l = 0; l < CHUNK; ++l) {
    a += dA[(bc * CHUNK + l) * NHEADS + h];
    Acs[e * CHUNK + l] = a;
  }
}

// ------------------------------------------------------- G = C * B^T per (b,c): fp32 WMMA 16x16x4
__global__ __launch_bounds__(32)
void k_G(const float* __restrict__ xBC, float* __restrict__ G) {
  int lane = threadIdx.x, r = lane & 15, hf = lane >> 4;
  int bc = blockIdx.x;
  int ti = blockIdx.y >> 2, tj = blockIdx.y & 3;
  int l0 = ti * 16, s0 = tj * 16;
  const float* crow = xBC + (size_t)(bc * CHUNK + l0 + r) * CONVDIM + DINNER + DSTATE; // Cm
  const float* brow = xBC + (size_t)(bc * CHUNK + s0 + r) * CONVDIM + DINNER;          // Bm
  v8f acc = {};
  for (int k = 0; k < DSTATE; k += 4) {
    v2f av = *(const v2f*)(crow + k + 2 * hf);
    v2f bv = *(const v2f*)(brow + k + 2 * hf);
    acc = __builtin_amdgcn_wmma_f32_16x16x4_f32(false, av, false, bv, (short)0, acc, false, false);
  }
  float* gout = G + (size_t)bc * CHUNK * CHUNK;
  #pragma unroll
  for (int j = 0; j < 8; ++j)
    gout[(l0 + hf * 8 + j) * CHUNK + s0 + r] = acc[j];
}

// ------------------------------------------------------- chunk states: states[b,c,h,p,n]
__global__ void k_states(const float* __restrict__ xBC, const float* __restrict__ xdt,
                         const float* __restrict__ Acs, float* __restrict__ states) {
  __shared__ float w[CHUNK];
  int bch = blockIdx.x;                // bc*16 + h
  int bc = bch >> 4, h = bch & 15;
  int tid = threadIdx.x;
  const float* acs = Acs + bch * CHUNK;
  if (tid < CHUNK) w[tid] = expf(acs[CHUNK - 1] - acs[tid]);
  __syncthreads();
  for (int o = tid; o < HEADDIM * DSTATE; o += 256) {
    int pp = o >> 7, n = o & 127;
    float s = 0.0f;
    for (int l = 0; l < CHUNK; ++l) {
      int t = bc * CHUNK + l;
      s += xBC[(size_t)t * CONVDIM + DINNER + n] * w[l] *
           xdt[(size_t)t * DINNER + h * HEADDIM + pp];
    }
    states[(size_t)bch * HEADDIM * DSTATE + o] = s;
  }
}

// inter-chunk scan: in-place convert per-chunk states -> prefix states (state BEFORE chunk c)
__global__ void k_scan(float* __restrict__ states, const float* __restrict__ Acs) {
  int bh = blockIdx.x;                 // b*16 + h
  int b = bh >> 4, h = bh & 15;
  int tid = threadIdx.x;
  float dec[NCHUNK];
  #pragma unroll
  for (int c = 0; c < NCHUNK; ++c)
    dec[c] = expf(Acs[((b * NCHUNK + c) * NHEADS + h) * CHUNK + CHUNK - 1]);
  for (int o = tid; o < HEADDIM * DSTATE; o += 256) {
    float S = 0.0f;
    #pragma unroll
    for (int c = 0; c < NCHUNK; ++c) {
      size_t idx = (size_t)((b * NCHUNK + c) * NHEADS + h) * HEADDIM * DSTATE + o;
      float tmp = states[idx];
      states[idx] = S;
      S = S * dec[c] + tmp;
    }
  }
}

// ------------------------------------------------------- Y = diag + off + D*x (fp32 WMMA)
__global__ __launch_bounds__(32)
void k_Y(const float* __restrict__ G, const float* __restrict__ xdt,
         const float* __restrict__ xBC, const float* __restrict__ Acs,
         const float* __restrict__ pref, const float* __restrict__ Dp,
         float* __restrict__ Y) {
  int lane = threadIdx.x, r = lane & 15, hf = lane >> 4;
  int bch = blockIdx.x, bc = bch >> 4, h = bch & 15;
  int ti = blockIdx.y >> 2, tj = blockIdx.y & 3;
  int l0 = ti * 16, p0 = tj * 16;
  int lrow = l0 + r, pcol = p0 + r;
  const float* acs = Acs + bch * CHUNK;
  float aL = acs[lrow];
  float eL = expf(aL);
  const float* grow = G + (size_t)bc * CHUNK * CHUNK + lrow * CHUNK;
  v8f acc = {};
  // diagonal (intra-chunk) part: A = (G .* Lm), B = xdt^T gathered per column p
  for (int k = 0; k < CHUNK; k += 4) {
    int s0i = k + 2 * hf;
    float a0 = (s0i     <= lrow) ? grow[s0i]     * expf(aL - acs[s0i])     : 0.0f;
    float a1 = (s0i + 1 <= lrow) ? grow[s0i + 1] * expf(aL - acs[s0i + 1]) : 0.0f;
    v2f av = {a0, a1};
    const float* xb = xdt + (size_t)(bc * CHUNK + s0i) * DINNER + h * HEADDIM + pcol;
    v2f bv = {xb[0], xb[DINNER]};
    acc = __builtin_amdgcn_wmma_f32_16x16x4_f32(false, av, false, bv, (short)0, acc, false, false);
  }
  // off-diagonal part: (C * e^{Acs}) @ prefix^T, prefix rows contiguous in n
  const float* crow = xBC + (size_t)(bc * CHUNK + lrow) * CONVDIM + DINNER + DSTATE;
  const float* prow = pref + (size_t)bch * HEADDIM * DSTATE + pcol * DSTATE;
  for (int k = 0; k < DSTATE; k += 4) {
    int n = k + 2 * hf;
    v2f av = {crow[n] * eL, crow[n + 1] * eL};
    v2f bv = *(const v2f*)(prow + n);
    acc = __builtin_amdgcn_wmma_f32_16x16x4_f32(false, av, false, bv, (short)0, acc, false, false);
  }
  float dph = Dp[h];
  #pragma unroll
  for (int j = 0; j < 8; ++j) {
    int l = l0 + hf * 8 + j;
    int t = bc * CHUNK + l;
    int ch = h * HEADDIM + p0 + r;
    Y[(size_t)t * DINNER + ch] = acc[j] + xBC[(size_t)t * CONVDIM + ch] * dph;
  }
}

// ------------------------------------------------------- gated rmsnorm -> bf16
__global__ void k_gatenorm(const float* __restrict__ Y, const float* __restrict__ zx,
                           const float* __restrict__ w, bf16* __restrict__ o) {
  __shared__ float sb[256];
  int t = blockIdx.x, tid = threadIdx.x;
  float g[4]; float ss = 0.0f;
  const float* yr = Y + (size_t)t * DINNER;
  const float* zr = zx + (size_t)t * DINPROJ;
  #pragma unroll
  for (int e = 0; e < 4; ++e) {
    int i = tid + e * 256;
    float z = zr[i];
    float gv = yr[i] * (z / (1.0f + expf(-z)));
    g[e] = gv; ss += gv * gv;
  }
  sb[tid] = ss; __syncthreads();
  for (int s = 128; s > 0; s >>= 1) { if (tid < s) sb[tid] += sb[tid + s]; __syncthreads(); }
  float ri = rsqrtf(sb[0] / (float)DINNER + RMS_EPS);
  bf16* orow = o + (size_t)t * DINNER;
  #pragma unroll
  for (int e = 0; e < 4; ++e) { int i = tid + e * 256; orow[i] = f2bf(g[e] * ri * w[i]); }
}

// ------------------------------------------------------- final norm factors, pool, head
__global__ void k_rinv(const float* __restrict__ x, float* __restrict__ rinv) {
  __shared__ float sb[256];
  int t = blockIdx.x, tid = threadIdx.x;
  const float* xr = x + (size_t)t * DMODEL;
  float v0 = xr[tid], v1 = xr[tid + 256];
  sb[tid] = v0 * v0 + v1 * v1; __syncthreads();
  for (int s = 128; s > 0; s >>= 1) { if (tid < s) sb[tid] += sb[tid + s]; __syncthreads(); }
  if (tid == 0) rinv[t] = rsqrtf(sb[0] / (float)DMODEL + RMS_EPS);
}

__global__ void k_pool(const float* __restrict__ x, const float* __restrict__ rinv,
                       const float* __restrict__ nfw, float* __restrict__ pooled) {
  int b = blockIdx.x;
  int d = blockIdx.y * 256 + threadIdx.x;
  float s = 0.0f;
  for (int l = 0; l < SEQLEN; ++l) {
    int t = b * SEQLEN + l;
    s += x[(size_t)t * DMODEL + d] * rinv[t];
  }
  pooled[b * DMODEL + d] = s * nfw[d] * (1.0f / (float)SEQLEN);
}

__global__ void k_cls(const float* __restrict__ pooled, const float* __restrict__ cw,
                      const float* __restrict__ cb, float* __restrict__ out) {
  int tid = threadIdx.x;
  if (tid >= BATCH * 2) return;
  int b = tid >> 1, c = tid & 1;
  float s = cb[c];
  for (int d = 0; d < DMODEL; ++d) s += pooled[b * DMODEL + d] * cw[c * DMODEL + d];
  out[tid] = s;
}

// ================================================================= launcher
extern "C" void kernel_launch(void* const* d_in, const int* in_sizes, int n_in,
                              void* d_out, int out_size, void* d_ws, size_t ws_size,
                              hipStream_t stream) {
  (void)in_sizes; (void)n_in; (void)out_size; (void)ws_size;
  const int*   tokens = (const int*)d_in[0];
  const float* emb    = (const float*)d_in[1];
  const float* in_w   = (const float*)d_in[2];
  const float* conv_w = (const float*)d_in[3];
  const float* conv_b = (const float*)d_in[4];
  const float* dt_b   = (const float*)d_in[5];
  const float* A_log  = (const float*)d_in[6];
  const float* D_par  = (const float*)d_in[7];
  const float* mnw    = (const float*)d_in[8];
  const float* out_w  = (const float*)d_in[9];
  const float* lnw    = (const float*)d_in[10];
  const float* nfw    = (const float*)d_in[11];
  const float* cls_w  = (const float*)d_in[12];
  const float* cls_b  = (const float*)d_in[13];
  float* out = (float*)d_out;

  char* p = (char*)d_ws;
  auto alloc = [&](size_t bytes) -> char* {
    char* r = p;
    p += (bytes + 255) & ~(size_t)255;
    return r;
  };
  float* x      = (float*)alloc((size_t)NT * DMODEL * 4);
  bf16*  wAb    = (bf16*) alloc((size_t)NLAYER * DINPROJ * DMODEL * 2);
  bf16*  wOb    = (bf16*) alloc((size_t)NLAYER * DMODEL * DINNER * 2);
  bf16*  xnb    = (bf16*) alloc((size_t)NT * DMODEL * 2);
  float* zx     = (float*)alloc((size_t)NT * DINPROJ * 4);
  float* xBC    = (float*)alloc((size_t)NT * CONVDIM * 4);
  float* dtbuf  = (float*)alloc((size_t)NT * NHEADS * 4);
  float* dAbuf  = (float*)alloc((size_t)NT * NHEADS * 4);
  float* xdt    = (float*)alloc((size_t)NT * DINNER * 4);
  float* Acs    = (float*)alloc((size_t)NT * NHEADS * 4);
  float* G      = (float*)alloc((size_t)BATCH * NCHUNK * CHUNK * CHUNK * 4);
  float* states = (float*)alloc((size_t)BATCH * NCHUNK * NHEADS * HEADDIM * DSTATE * 4);
  float* Y      = (float*)alloc((size_t)NT * DINNER * 4);
  bf16*  yb     = (bf16*) alloc((size_t)NT * DINNER * 2);
  float* rinv   = (float*)alloc((size_t)NT * 4);
  float* pooled = (float*)alloc((size_t)BATCH * DMODEL * 4);

  {
    int n1 = NLAYER * DINPROJ * DMODEL;
    k_f32_to_bf16<<<(n1 + 255) / 256, 256, 0, stream>>>(in_w, wAb, n1);
    int n2 = NLAYER * DMODEL * DINNER;
    k_f32_to_bf16<<<(n2 + 255) / 256, 256, 0, stream>>>(out_w, wOb, n2);
  }
  k_embed<<<NT, 256, 0, stream>>>(tokens, emb, x);

  for (int L = 0; L < NLAYER; ++L) {
    k_rmsnorm_in<<<NT, 256, 0, stream>>>(x, lnw + L * DMODEL, xnb);
    dim3 g1(NT / 64, (DINPROJ + 63) / 64);
    k_gemm_bf16<<<g1, 128, 0, stream>>>(xnb, wAb + (size_t)L * DINPROJ * DMODEL,
                                        zx, NT, DINPROJ, DMODEL, 0);
    k_dt<<<(NT * NHEADS) / 256, 256, 0, stream>>>(zx, dt_b + L * NHEADS,
                                                  A_log + L * NHEADS, dtbuf, dAbuf);
    dim3 gc(NT, CONVDIM / 256);
    k_conv<<<gc, 256, 0, stream>>>(zx, conv_w + (size_t)L * CONVDIM * 4,
                                   conv_b + L * CONVDIM, xBC);
    k_xdt<<<(NT * DINNER) / 256, 256, 0, stream>>>(xBC, dtbuf, xdt);
    k_cumsum<<<(BATCH * NCHUNK * NHEADS) / 256, 256, 0, stream>>>(dAbuf, Acs);
    dim3 gG(BATCH * NCHUNK, 16);
    k_G<<<gG, 32, 0, stream>>>(xBC, G);
    k_states<<<BATCH * NCHUNK * NHEADS, 256, 0, stream>>>(xBC, xdt, Acs, states);
    k_scan<<<BATCH * NHEADS, 256, 0, stream>>>(states, Acs);
    dim3 gY(BATCH * NCHUNK * NHEADS, 16);
    k_Y<<<gY, 32, 0, stream>>>(G, xdt, xBC, Acs, states, D_par + L * NHEADS, Y);
    k_gatenorm<<<NT, 256, 0, stream>>>(Y, zx, mnw + L * DINNER, yb);
    dim3 g2(NT / 64, DMODEL / 64);
    k_gemm_bf16<<<g2, 128, 0, stream>>>(yb, wOb + (size_t)L * DMODEL * DINNER,
                                        x, NT, DMODEL, DINNER, 1);
  }

  k_rinv<<<NT, 256, 0, stream>>>(x, rinv);
  dim3 gp(BATCH, DMODEL / 256);
  k_pool<<<gp, 256, 0, stream>>>(x, rinv, nfw, pooled);
  k_cls<<<1, 16, 0, stream>>>(pooled, cls_w, cls_b, out);
}
